// LocalDownsampleFlexAttnProcessor_43404939494161
// MI455X (gfx1250) — matrix operand order repaired and checked by reading, not compiled, not run on hardware
//
#include <hip/hip_runtime.h>
#include <hip/hip_fp16.h>

typedef _Float16 f16;
typedef _Float16 v16h __attribute__((ext_vector_type(16)));
typedef _Float16 v8h  __attribute__((ext_vector_type(8)));
typedef float    v8f  __attribute__((ext_vector_type(8)));
typedef int      v4i_t __attribute__((vector_size(16)));

#define TXT_N   512
#define WID_N   64
#define IMG_N   4096
#define SEQ_N   4608          // TXT + IMG
#define HEADS_N 8
#define HD_N    128
#define DM_N    1024
#define DOWN_N  256
#define SKV_N   4864          // SEQ + DOWN
#define RAD2    64            // RADIUS^2

#ifndef __has_builtin
#define __has_builtin(x) 0
#endif
#if __has_builtin(__builtin_amdgcn_global_load_async_to_lds_b128) && \
    __has_builtin(__builtin_amdgcn_s_wait_asynccnt)
#define HAVE_ASYNC_LDS 1
#else
#define HAVE_ASYNC_LDS 0
#endif

// ---------------------------------------------------------------- fragments
static __device__ __forceinline__ v16h mk16(v8h a, v8h b){
  v16h r;
#pragma unroll
  for (int i = 0; i < 8; ++i){ r[i] = a[i]; r[i+8] = b[i]; }
  return r;
}

// A-fragment (16xK slab, row-major, 16-bit): lane holds row M=lane%16,
// elements 0..15 -> K = 16*half+e (contiguous).
static __device__ __forceinline__ v16h load_afrag(const f16* base, int stride,
                                                  int row0, int col0){
  int lane = threadIdx.x & 31, hf = lane >> 4, m = lane & 15;
  const f16* p = base + (size_t)(row0 + m) * stride + col0 + hf * 16;
  v8h lo = *(const v8h*)p;
  v8h hi = *(const v8h*)(p + 8);
  return mk16(lo, hi);
}

// B-fragment via B^T (row-major [N][K]): lane holds column N=lane%16,
// elements = K = 16*half + 0..15 (one contiguous 32B run).
static __device__ __forceinline__ v16h load_bfrag(const f16* baseT, int stride,
                                                  int n0, int col0){
  int lane = threadIdx.x & 31, hf = lane >> 4, m = lane & 15;
  const f16* p = baseT + (size_t)(n0 + m) * stride + col0 + hf * 16;
  v8h lo = *(const v8h*)p;
  v8h hi = *(const v8h*)(p + 8);
  return mk16(lo, hi);
}

// PV B-fragment from Vt[h][d][SKV]; local K 0..15 -> keys j0+e, 16..31 -> j1+e.
static __device__ __forceinline__ v16h load_pv_bfrag(const f16* Vth, int d0,
                                                     int j0, int j1){
  int lane = threadIdx.x & 31, hf = lane >> 4, m = lane & 15;
  const f16* p = Vth + (size_t)(d0 + m) * SKV_N + (hf ? j1 : j0);
  v8h lo = *(const v8h*)p;
  v8h hi = *(const v8h*)(p + 8);
  return mk16(lo, hi);
}

static __device__ __forceinline__ v8f wmma32(v16h a, v16h b, v8f c){
  return __builtin_amdgcn_wmma_f32_16x16x32_f16(false, a, false, b,
                                                (short)0, c, false, false);
}

static __device__ __forceinline__ v8f vzero8(){
  v8f z = {0.f,0.f,0.f,0.f,0.f,0.f,0.f,0.f};
  return z;
}

// Stage one 16x32 f16 A slab (1 KB) into LDS: 64 lanes x 16B each.
// Uses the CDNA5 async global->LDS path when the toolchain exposes it.
static __device__ __forceinline__ void stage_a(f16* dst, const f16* src,
                                               int m0, int k0, int stride,
                                               int tid){
  if (tid < 64){
    int row = tid >> 2, c = (tid & 3) * 8;
    const f16* g = src + (size_t)(m0 + row) * stride + k0 + c;
    f16* l = dst + row * 32 + c;
#if HAVE_ASYNC_LDS
    __builtin_amdgcn_global_load_async_to_lds_b128(
        (__attribute__((address_space(1))) v4i_t*)(g),
        (__attribute__((address_space(3))) v4i_t*)(l), 0, 0);
#else
    *(v8h*)l = *(const v8h*)g;
#endif
  }
}

static __device__ __forceinline__ void join_block(){
#if HAVE_ASYNC_LDS
  __builtin_amdgcn_s_wait_asynccnt(0);
#endif
  __syncthreads();
}

// ---------------------------------------------------------------- K1: pack X
__global__ void k_pack_x(const float* __restrict__ hid,
                         const float* __restrict__ enc, f16* __restrict__ X16){
  int i = blockIdx.x * blockDim.x + threadIdx.x;
  if (i >= SEQ_N * DM_N) return;
  int s = i >> 10, c = i & 1023;
  float v = (s < TXT_N) ? enc[s * DM_N + c] : hid[(s - TXT_N) * DM_N + c];
  X16[i] = (f16)v;
}

// ----------------------------------------------------- K2: W[k][n] -> Wt[n][k]
__global__ void k_transpose_w(const float* __restrict__ W, f16* __restrict__ Wt){
  int i = blockIdx.x * blockDim.x + threadIdx.x;
  if (i >= DM_N * DM_N) return;
  int o = i >> 10, k = i & 1023;
  Wt[i] = (f16)W[k * DM_N + o];
}

// ---------------------------------------------- K3: QKV projection via WMMA
// One block = 16 M-rows x 256 N-cols: 4 waves x 4 N-tiles, LDS-shared A slab
// (double-buffered via async global->LDS when available).
// mode 0: Q[h][SEQ][128]; mode 1: K rows of Kf[h][SKV][128]; mode 2: Vt[h][128][SKV]
__global__ __launch_bounds__(128)
void k_proj(const f16* __restrict__ X,
            const f16* __restrict__ WtTxt, const float* __restrict__ bTxt,
            const f16* __restrict__ WtImg, const float* __restrict__ bImg,
            f16* __restrict__ out, int mode){
  __shared__ alignas(16) f16 Abuf[2][16 * 32];
  int tid  = threadIdx.x;
  int wave = tid >> 5;
  int lane = tid & 31, hf = lane >> 4, m = lane & 15;
  int m0 = blockIdx.x * 16;                    // 288 row tiles
  int ntBase = blockIdx.y * 16 + wave * 4;     // 64 n-tiles, 4 per wave
  const f16*   Wt   = (m0 < TXT_N) ? WtTxt : WtImg;
  const float* bias = (m0 < TXT_N) ? bTxt : bImg;

  v8f acc[4];
#pragma unroll
  for (int t = 0; t < 4; ++t) acc[t] = vzero8();

  stage_a(Abuf[0], X, m0, 0, DM_N, tid);
  join_block();

  for (int k0 = 0; k0 < DM_N; k0 += 32){
    int cur = (k0 >> 5) & 1;
    if (k0 + 32 < DM_N) stage_a(Abuf[cur ^ 1], X, m0, k0 + 32, DM_N, tid);
    v16h a = load_afrag(Abuf[cur], 32, 0, 0);
#pragma unroll
    for (int t = 0; t < 4; ++t){
      v16h b = load_bfrag(Wt, DM_N, (ntBase + t) * 16, k0);
      acc[t] = wmma32(a, b, acc[t]);
    }
    join_block();
  }

#pragma unroll
  for (int t = 0; t < 4; ++t){
    int n = (ntBase + t) * 16 + m, h = n >> 7, d = n & 127;
    float bv = bias[n];
    if (mode == 2){
      v8h st;
#pragma unroll
      for (int r = 0; r < 8; ++r) st[r] = (f16)(acc[t][r] + bv);
      f16* p = out + (size_t)(h * HD_N + d) * SKV_N + m0 + hf * 8;
      *(v8h*)p = st;
    } else {
      int rows = (mode == 1) ? SKV_N : SEQ_N;
#pragma unroll
      for (int r = 0; r < 8; ++r){
        int s = m0 + hf * 8 + r;
        out[(size_t)(h * rows + s) * HD_N + d] = (f16)(acc[t][r] + bv);
      }
    }
  }
}

// ---------------------------------------------- K4: RMSNorm + RoPE on Q and K
static __device__ __forceinline__ void normrope_row(f16* row, const float* w,
                                                    const float* ct,
                                                    const float* st, int lane){
  int e = lane * 4;
  float x0 = (float)row[e],   x1 = (float)row[e+1];
  float x2 = (float)row[e+2], x3 = (float)row[e+3];
  float ss = x0*x0 + x1*x1 + x2*x2 + x3*x3;
  ss += __shfl_xor(ss, 1, 32);  ss += __shfl_xor(ss, 2, 32);
  ss += __shfl_xor(ss, 4, 32);  ss += __shfl_xor(ss, 8, 32);
  ss += __shfl_xor(ss, 16, 32);
  float inv = rsqrtf(ss * (1.0f / HD_N) + 1e-6f);
  float n0 = x0*inv*w[e],   n1 = x1*inv*w[e+1];
  float n2 = x2*inv*w[e+2], n3 = x3*inv*w[e+3];
  float c0 = ct[e], s0 = st[e], c1 = ct[e+2], s1 = st[e+2];
  row[e]   = (f16)(n0*c0 - n1*s0);
  row[e+1] = (f16)(n1*c0 + n0*s0);
  row[e+2] = (f16)(n2*c1 - n3*s1);
  row[e+3] = (f16)(n3*c1 + n2*s1);
}

__global__ __launch_bounds__(32)
void k_norm_rope(f16* __restrict__ Q, f16* __restrict__ K,
                 const float* __restrict__ nq,  const float* __restrict__ nk,
                 const float* __restrict__ nqa, const float* __restrict__ nka,
                 const float* __restrict__ cosT, const float* __restrict__ sinT){
  int s = blockIdx.x, h = blockIdx.y, lane = threadIdx.x;
  const float* ct = cosT + (size_t)s * HD_N;
  const float* st = sinT + (size_t)s * HD_N;
  normrope_row(Q + (size_t)(h * SEQ_N + s) * HD_N, (s < TXT_N) ? nqa : nq, ct, st, lane);
  normrope_row(K + (size_t)(h * SKV_N + s) * HD_N, (s < TXT_N) ? nka : nk, ct, st, lane);
}

// ---------------------------------------------- K5: 4x4 weighted-pool downsample
__global__ __launch_bounds__(128)
void k_down_k(f16* __restrict__ Kf, const float* __restrict__ sw){
  int p = blockIdx.x, h = blockIdx.y, d = threadIdx.x;
  int ph = p >> 4, pw = p & 15;
  float acc = 0.f;
#pragma unroll
  for (int i = 0; i < 4; ++i)
#pragma unroll
    for (int j = 0; j < 4; ++j){
      int s = TXT_N + (ph*4 + i) * WID_N + pw*4 + j;
      acc += (float)Kf[(size_t)(h * SKV_N + s) * HD_N + d] * sw[i*4 + j];
    }
  Kf[(size_t)(h * SKV_N + SEQ_N + p) * HD_N + d] = (f16)acc;
}

__global__ __launch_bounds__(256)
void k_down_v(f16* __restrict__ Vt, const float* __restrict__ sw){
  int d = blockIdx.x, h = blockIdx.y, p = threadIdx.x;
  int ph = p >> 4, pw = p & 15;
  f16* row = Vt + (size_t)(h * HD_N + d) * SKV_N;
  float acc = 0.f;
#pragma unroll
  for (int i = 0; i < 4; ++i)
#pragma unroll
    for (int j = 0; j < 4; ++j)
      acc += (float)row[TXT_N + (ph*4 + i) * WID_N + pw*4 + j] * sw[i*4 + j];
  row[SEQ_N + p] = (f16)acc;
}

// ---------------------------------------------- K6: masked flash attention
static __device__ __forceinline__ void attn_chunk(
    const v16h* qf, const f16* Kh, const f16* Vth, f16* pb,
    int j0, int j1, bool use_mask, int dh2, int c0, int wk0, int wk1,
    float* mi, float* li, v8f* oacc)
{
  const float scale = 0.08838834764831845f;   // 128^-0.5
  int lane = threadIdx.x & 31, hf = lane >> 4, m = lane & 15;

  __builtin_prefetch(Kh + (size_t)(j0 + 32) * HD_N, 0, 1);

  v8f s0 = vzero8(), s1 = vzero8();
#pragma unroll
  for (int kk = 0; kk < 4; ++kk)
    s0 = wmma32(qf[kk], load_bfrag(Kh, HD_N, j0, kk * 32), s0);
  if (j1 >= 0){
#pragma unroll
    for (int kk = 0; kk < 4; ++kk)
      s1 = wmma32(qf[kk], load_bfrag(Kh, HD_N, j1, kk * 32), s1);
  }

  float p0[8], p1[8];
#pragma unroll
  for (int r = 0; r < 8; ++r){
    float v0 = s0[r] * scale, v1 = s1[r] * scale;
    if (use_mask){
      int wq  = c0 + r + 8 * hf;
      int dw0 = wq - (wk0 + m), dw1 = wq - (wk1 + m);
      if (dh2 + dw0 * dw0 > RAD2) v0 = -INFINITY;
      if (dh2 + dw1 * dw1 > RAD2) v1 = -INFINITY;
    }
    if (j1 < 0) v1 = -INFINITY;
    p0[r] = v0; p1[r] = v1;
  }

  float mnew[8], alpha[8];
#pragma unroll
  for (int r = 0; r < 8; ++r){
    float t = fmaxf(p0[r], p1[r]);
    t = fmaxf(t, __shfl_xor(t, 1, 32));
    t = fmaxf(t, __shfl_xor(t, 2, 32));
    t = fmaxf(t, __shfl_xor(t, 4, 32));
    t = fmaxf(t, __shfl_xor(t, 8, 32));
    float mn = fmaxf(mi[r], t);
    alpha[r] = __expf(mi[r] - mn);
    mnew[r]  = mn;
  }
#pragma unroll
  for (int r = 0; r < 8; ++r){
    float e0 = __expf(p0[r] - mnew[r]);
    float e1 = __expf(p1[r] - mnew[r]);
    float rs = e0 + e1;
    rs += __shfl_xor(rs, 1, 32);  rs += __shfl_xor(rs, 2, 32);
    rs += __shfl_xor(rs, 4, 32);  rs += __shfl_xor(rs, 8, 32);
    li[r] = li[r] * alpha[r] + rs;
    mi[r] = mnew[r];
    int row = r + 8 * hf;
    pb[row * 32 + m]      = (f16)e0;
    pb[row * 32 + 16 + m] = (f16)e1;
  }
#pragma unroll
  for (int dc = 0; dc < 8; ++dc)
#pragma unroll
    for (int r = 0; r < 8; ++r) oacc[dc][r] *= alpha[r];

  v16h pf = load_afrag(pb, 32, 0, 0);
  int j1v = (j1 >= 0) ? j1 : j0;
#pragma unroll
  for (int dc = 0; dc < 8; ++dc)
    oacc[dc] = wmma32(pf, load_pv_bfrag(Vth, dc * 16, j0, j1v), oacc[dc]);
}

__global__ __launch_bounds__(128)
void k_attn(const f16* __restrict__ Q, const f16* __restrict__ Kf,
            const f16* __restrict__ Vt, f16* __restrict__ O16){
  __shared__ alignas(16) f16 Pbuf[4][16 * 32];
  int wave = threadIdx.x >> 5;
  int lane = threadIdx.x & 31, hf = lane >> 4, m = lane & 15;
  int tq = blockIdx.x * 4 + wave;           // 0..287
  int h  = blockIdx.y;
  int qs = tq * 16;
  const f16* Qh  = Q  + (size_t)h * SEQ_N * HD_N;
  const f16* Kh  = Kf + (size_t)h * SKV_N * HD_N;
  const f16* Vth = Vt + (size_t)h * HD_N * SKV_N;
  f16* pb = Pbuf[wave];

  v16h qf[4];
#pragma unroll
  for (int kk = 0; kk < 4; ++kk) qf[kk] = load_afrag(Qh, HD_N, qs, kk * 32);

  float mi[8], li[8]; v8f oacc[8];
#pragma unroll
  for (int r = 0; r < 8; ++r){ mi[r] = -INFINITY; li[r] = 0.f; }
#pragma unroll
  for (int dc = 0; dc < 8; ++dc) oacc[dc] = vzero8();

  if (qs < TXT_N){
    for (int j0 = 0; j0 < SEQ_N; j0 += 32)
      attn_chunk(qf, Kh, Vth, pb, j0, j0 + 16, false, 0, 0, 0, 0, mi, li, oacc);
  } else {
    int iq = qs - TXT_N, hq = iq >> 6, c0 = iq & 63;
    for (int j0 = 0; j0 < TXT_N; j0 += 32)                       // all text keys
      attn_chunk(qf, Kh, Vth, pb, j0, j0 + 16, false, 0, 0, 0, 0, mi, li, oacc);
    for (int j0 = SEQ_N; j0 < SKV_N; j0 += 32)                   // all down keys
      attn_chunk(qf, Kh, Vth, pb, j0, j0 + 16, false, 0, 0, 0, 0, mi, li, oacc);
    int hk0 = (hq > 8) ? hq - 8 : 0;
    int hk1 = (hq + 8 < WID_N - 1) ? hq + 8 : WID_N - 1;
    int ct   = c0 >> 4;
    int t_lo = (ct > 0) ? ct - 1 : 0;
    int t_hi = (ct < 3) ? ct + 1 : 3;
    for (int hk = hk0; hk <= hk1; ++hk){
      int dh = hq - hk, dh2 = dh * dh;
      int rowbase = TXT_N + hk * WID_N;
      for (int t = t_lo; t <= t_hi; t += 2){
        int j0 = rowbase + t * 16;
        int j1 = (t + 1 <= t_hi) ? rowbase + (t + 1) * 16 : -1;
        attn_chunk(qf, Kh, Vth, pb, j0, j1, true, dh2, c0, t * 16, (t + 1) * 16,
                   mi, li, oacc);
      }
    }
  }

  float inv[8];
#pragma unroll
  for (int r = 0; r < 8; ++r) inv[r] = 1.f / li[r];
#pragma unroll
  for (int dc = 0; dc < 8; ++dc)
#pragma unroll
    for (int r = 0; r < 8; ++r){
      int s = qs + r + 8 * hf;
      int col = h * HD_N + dc * 16 + m;
      O16[(size_t)s * DM_N + col] = (f16)(oacc[dc][r] * inv[r]);
    }
}

// ---------------------------------------------- K7: output projections
__global__ __launch_bounds__(128)
void k_outproj(const f16* __restrict__ O16,
               const f16* __restrict__ WtOut, const float* __restrict__ bOut,
               const f16* __restrict__ WtAdd, const float* __restrict__ bAdd,
               float* __restrict__ out){
  __shared__ alignas(16) f16 Abuf[2][16 * 32];
  int tid  = threadIdx.x;
  int wave = tid >> 5;
  int lane = tid & 31, hf = lane >> 4, m = lane & 15;
  int m0 = blockIdx.x * 16;
  int ntBase = blockIdx.y * 16 + wave * 4;
  const f16*   Wt  = (m0 < TXT_N) ? WtAdd : WtOut;
  const float* bia = (m0 < TXT_N) ? bAdd  : bOut;

  v8f acc[4];
#pragma unroll
  for (int t = 0; t < 4; ++t) acc[t] = vzero8();

  stage_a(Abuf[0], O16, m0, 0, DM_N, tid);
  join_block();

  for (int k0 = 0; k0 < DM_N; k0 += 32){
    int cur = (k0 >> 5) & 1;
    if (k0 + 32 < DM_N) stage_a(Abuf[cur ^ 1], O16, m0, k0 + 32, DM_N, tid);
    v16h a = load_afrag(Abuf[cur], 32, 0, 0);
#pragma unroll
    for (int t = 0; t < 4; ++t){
      v16h b = load_bfrag(Wt, DM_N, (ntBase + t) * 16, k0);
      acc[t] = wmma32(a, b, acc[t]);
    }
    join_block();
  }

#pragma unroll
  for (int t = 0; t < 4; ++t){
    int n = (ntBase + t) * 16 + m;
    float bv = bia[n];
#pragma unroll
    for (int r = 0; r < 8; ++r){
      int s = m0 + hf * 8 + r;
      size_t base = (s < TXT_N) ? ((size_t)IMG_N * DM_N + (size_t)s * DM_N)
                                : ((size_t)(s - TXT_N) * DM_N);
      out[base + n] = acc[t][r] + bv;
    }
  }
}

// ---------------------------------------------------------------- launch
extern "C" void kernel_launch(void* const* d_in, const int* in_sizes, int n_in,
                              void* d_out, int out_size, void* d_ws, size_t ws_size,
                              hipStream_t stream){
  (void)in_sizes; (void)n_in; (void)out_size; (void)ws_size;
  const float* hid  = (const float*)d_in[0];
  const float* enc  = (const float*)d_in[1];
  const float* Wq   = (const float*)d_in[2];  const float* bq   = (const float*)d_in[3];
  const float* Wk   = (const float*)d_in[4];  const float* bk   = (const float*)d_in[5];
  const float* Wv   = (const float*)d_in[6];  const float* bv   = (const float*)d_in[7];
  const float* Wqa  = (const float*)d_in[8];  const float* bqa  = (const float*)d_in[9];
  const float* Wka  = (const float*)d_in[10]; const float* bka  = (const float*)d_in[11];
  const float* Wva  = (const float*)d_in[12]; const float* bva  = (const float*)d_in[13];
  const float* nq   = (const float*)d_in[14]; const float* nk   = (const float*)d_in[15];
  const float* nqa  = (const float*)d_in[16]; const float* nka  = (const float*)d_in[17];
  const float* sw   = (const float*)d_in[18];
  const float* rc   = (const float*)d_in[19]; const float* rs   = (const float*)d_in[20];
  const float* Wout = (const float*)d_in[21]; const float* bout = (const float*)d_in[22];
  const float* Wadd = (const float*)d_in[23]; const float* badd = (const float*)d_in[24];
  float* out = (float*)d_out;

  const size_t NW = (size_t)DM_N * DM_N;
  f16* X16  = (f16*)d_ws;
  f16* WtQ  = X16  + (size_t)SEQ_N * DM_N;
  f16* WtK  = WtQ  + NW;
  f16* WtV  = WtK  + NW;
  f16* WtQa = WtV  + NW;
  f16* WtKa = WtQa + NW;
  f16* WtVa = WtKa + NW;
  f16* WtO  = WtVa + NW;
  f16* WtA  = WtO  + NW;
  f16* Qb   = WtA  + NW;
  f16* Kf   = Qb   + (size_t)HEADS_N * SEQ_N * HD_N;
  f16* Vt   = Kf   + (size_t)HEADS_N * SKV_N * HD_N;
  f16* O16  = Vt   + (size_t)HEADS_N * SKV_N * HD_N;   // ~65 MB total in d_ws

  k_pack_x<<<(SEQ_N * DM_N + 255) / 256, 256, 0, stream>>>(hid, enc, X16);

  const int tg = (DM_N * DM_N + 255) / 256;
  k_transpose_w<<<tg, 256, 0, stream>>>(Wq,   WtQ);
  k_transpose_w<<<tg, 256, 0, stream>>>(Wk,   WtK);
  k_transpose_w<<<tg, 256, 0, stream>>>(Wv,   WtV);
  k_transpose_w<<<tg, 256, 0, stream>>>(Wqa,  WtQa);
  k_transpose_w<<<tg, 256, 0, stream>>>(Wka,  WtKa);
  k_transpose_w<<<tg, 256, 0, stream>>>(Wva,  WtVa);
  k_transpose_w<<<tg, 256, 0, stream>>>(Wout, WtO);
  k_transpose_w<<<tg, 256, 0, stream>>>(Wadd, WtA);

  dim3 gp(SEQ_N / 16, 4);             // 288 x 4, 4 waves x 4 N-tiles per block
  k_proj<<<gp, 128, 0, stream>>>(X16, WtQa, bqa, WtQ, bq, Qb, 0);
  k_proj<<<gp, 128, 0, stream>>>(X16, WtKa, bka, WtK, bk, Kf, 1);
  k_proj<<<gp, 128, 0, stream>>>(X16, WtVa, bva, WtV, bv, Vt, 2);

  k_norm_rope<<<dim3(SEQ_N, HEADS_N), 32, 0, stream>>>(Qb, Kf, nq, nk, nqa, nka, rc, rs);

  k_down_k<<<dim3(DOWN_N, HEADS_N), 128, 0, stream>>>(Kf, sw);
  k_down_v<<<dim3(HD_N,  HEADS_N), 256, 0, stream>>>(Vt, sw);

  k_attn<<<dim3(SEQ_N / 16 / 4, HEADS_N), 128, 0, stream>>>(Qb, Kf, Vt, O16);

  k_outproj<<<gp, 128, 0, stream>>>(O16, WtO, bout, WtA, badd, out);
}
